// N3StageBlock_27874337751172
// MI455X (gfx1250) — matrix operand order repaired and checked by reading, not compiled, not run on hardware
//
#include <hip/hip_runtime.h>
#include <hip/hip_bf16.h>

// ---------------------------------------------------------------------------
// CDNA5 async global->LDS copy availability (device pass only)
// ---------------------------------------------------------------------------
#ifdef __has_builtin
#  if __has_builtin(__builtin_amdgcn_global_load_async_to_lds_b128) && \
      __has_builtin(__builtin_amdgcn_s_wait_asynccnt)
#    define USE_ASYNC_LDS 1
#  endif
#endif
#ifndef USE_ASYNC_LDS
#  define USE_ASYNC_LDS 0
#endif

// ---------------------------------------------------------------------------
// Types for CDNA5 WMMA
// ---------------------------------------------------------------------------
typedef __attribute__((ext_vector_type(16))) __bf16 v16bf;
typedef __attribute__((ext_vector_type(8)))  float  v8f;
typedef __attribute__((ext_vector_type(4)))  int    v4i;

union frag_u {
    uint4 q[2];
    v16bf v;
};

static __device__ __forceinline__ __bf16 f2bf(float f) {
    unsigned u = __float_as_uint(f);
    u += 0x7fffu + ((u >> 16) & 1u);          // round-to-nearest-even
    unsigned short s = (unsigned short)(u >> 16);
    union { unsigned short u16; __bf16 b; } cv;
    cv.u16 = s;
    return cv.b;
}

static __device__ __forceinline__ float gelu_tanh(float x) {
    float x3 = x * x * x;
    return 0.5f * x * (1.0f + tanhf(0.7978845608028654f * (x + 0.044715f * x3)));
}

#if USE_ASYNC_LDS
typedef __attribute__((address_space(1))) v4i g_v4i;   // global
typedef __attribute__((address_space(3))) v4i l_v4i;   // LDS
static __device__ __forceinline__ void async_copy16(const __bf16* g, __bf16* l) {
    __builtin_amdgcn_global_load_async_to_lds_b128(
        (g_v4i*)g, (l_v4i*)l, 0, 0);
}
#endif

#define NTOK 4096
#define DMODEL 512
#define NEXP 8
#define HEXP 1024
#define DFF 2048

// ---------------------------------------------------------------------------
// Per-token prep: LN (both branches), feat/FiLM -> x_in, router top-2 gates,
// out init = hidden + alpha * sum_e gate_e * b2_e
// ---------------------------------------------------------------------------
__global__ __launch_bounds__(256) void prep_kernel(
    const float* __restrict__ hidden, const float* __restrict__ raw,
    const float* __restrict__ ln_g, const float* __restrict__ ln_b,
    const float* __restrict__ pre_g, const float* __restrict__ pre_b,
    const float* __restrict__ feat_w, const float* __restrict__ feat_b,
    const float* __restrict__ film_w, const float* __restrict__ film_b,
    const float* __restrict__ router_w, const float* __restrict__ router_b,
    const float* __restrict__ b2, const float* __restrict__ alpha_p,
    __bf16* __restrict__ xs_bf, __bf16* __restrict__ xin_bf,
    float* __restrict__ gate8, float* __restrict__ out)
{
    const int t = blockIdx.x;
    const int tid = threadIdx.x;
    __shared__ float red[256], red2[256];
    __shared__ float xbuf[DMODEL];
    __shared__ float feat_s[64], raw_s[16], logit_s[8], gates_s[8];

    const float* hrow = hidden + (size_t)t * DMODEL;
    float h0 = hrow[tid], h1 = hrow[tid + 256];
    red[tid]  = h0 + h1;
    red2[tid] = h0 * h0 + h1 * h1;
    __syncthreads();
    for (int s = 128; s > 0; s >>= 1) {
        if (tid < s) { red[tid] += red[tid + s]; red2[tid] += red2[tid + s]; }
        __syncthreads();
    }
    const float mean = red[0] * (1.0f / DMODEL);
    const float var  = red2[0] * (1.0f / DMODEL) - mean * mean;
    const float rstd = rsqrtf(var + 1e-5f);

    {
        int d0 = tid, d1 = tid + 256;
        float xn0 = (h0 - mean) * rstd, xn1 = (h1 - mean) * rstd;
        xbuf[d0] = xn0 * ln_g[d0] + ln_b[d0];
        xbuf[d1] = xn1 * ln_g[d1] + ln_b[d1];
        xs_bf[(size_t)t * DMODEL + d0] = f2bf(xn0 * pre_g[d0] + pre_b[d0]);
        xs_bf[(size_t)t * DMODEL + d1] = f2bf(xn1 * pre_g[d1] + pre_b[d1]);
    }
    if (tid < 16) raw_s[tid] = raw[(size_t)t * 16 + tid];
    __syncthreads();

    if (tid < 64) {
        float acc = feat_b[tid];
        #pragma unroll
        for (int i = 0; i < 16; ++i) acc += raw_s[i] * feat_w[i * 64 + tid];
        feat_s[tid] = acc;
    }
    __syncthreads();

    for (int d = tid; d < DMODEL; d += 256) {
        float g = film_b[d], bb = film_b[DMODEL + d];
        #pragma unroll 8
        for (int i = 0; i < 64; ++i) {
            float f = feat_s[i];
            g  += f * film_w[i * (2 * DMODEL) + d];
            bb += f * film_w[i * (2 * DMODEL) + DMODEL + d];
        }
        xin_bf[(size_t)t * DMODEL + d] = f2bf(xbuf[d] * (1.0f + g) + bb);
    }

    if (tid < 8) {
        float l = router_b[tid];
        for (int d = 0; d < DMODEL; ++d) l += xbuf[d] * router_w[d * NEXP + tid];
        logit_s[tid] = l;
    }
    __syncthreads();

    if (tid == 0) {
        int i1 = 0;
        for (int e = 1; e < 8; ++e) if (logit_s[e] > logit_s[i1]) i1 = e;
        int i2 = (i1 == 0) ? 1 : 0;
        for (int e = 0; e < 8; ++e) if (e != i1 && logit_s[e] > logit_s[i2]) i2 = e;
        float e2 = expf(logit_s[i2] - logit_s[i1]);
        float denom = 1.0f + e2;
        for (int e = 0; e < 8; ++e) gates_s[e] = 0.0f;
        gates_s[i1] = 1.0f / denom;
        gates_s[i2] = e2 / denom;
        for (int e = 0; e < 8; ++e) gate8[(size_t)t * NEXP + e] = gates_s[e];
    }
    __syncthreads();

    const float a = *alpha_p;
    {
        int d0 = tid, d1 = tid + 256;
        float s0 = 0.0f, s1 = 0.0f;
        #pragma unroll
        for (int e = 0; e < 8; ++e) {
            float ge = gates_s[e];
            s0 += ge * b2[e * DMODEL + d0];
            s1 += ge * b2[e * DMODEL + d1];
        }
        out[(size_t)t * DMODEL + d0] = h0 + a * s0;
        out[(size_t)t * DMODEL + d1] = h1 + a * s1;
    }
}

// ---------------------------------------------------------------------------
// Transpose + f32->bf16 convert: src[K,N] f32 (per blockIdx.z matrix) ->
// dst[N,K] bf16. LDS 32x32 tile, coalesced reads and writes.
// ---------------------------------------------------------------------------
__global__ __launch_bounds__(256) void transpose_cvt_kernel(
    const float* __restrict__ src, __bf16* __restrict__ dst, int K, int N)
{
    __shared__ float tile[32][33];
    const size_t mat = (size_t)blockIdx.z * (size_t)K * (size_t)N;
    const int tx = threadIdx.x;          // 0..31
    const int ty = threadIdx.y;          // 0..7
    const int k0 = blockIdx.y * 32;
    const int n0 = blockIdx.x * 32;
    #pragma unroll
    for (int i = 0; i < 4; ++i) {
        int k = k0 + ty + i * 8;
        tile[ty + i * 8][tx] = src[mat + (size_t)k * N + n0 + tx];
    }
    __syncthreads();
    #pragma unroll
    for (int i = 0; i < 4; ++i) {
        int n = n0 + ty + i * 8;
        dst[mat + (size_t)n * K + k0 + tx] = f2bf(tile[tx][ty + i * 8]);
    }
}

// ---------------------------------------------------------------------------
// bf16 WMMA GEMM: C[M,N] = epilogue(A[M,K] * B[K,N]), with B supplied
// TRANSPOSED as Bt[N,K] so every LDS access is a contiguous b128.
// Block tile 64x128, BK=32, 256 threads = 8 waves as 4(M) x 2(N);
// each wave computes 16x64 via 4 WMMA accumulators reusing one A fragment.
// Double-buffered LDS: one barrier per K-step; next tile staged (async
// direct-to-LDS when available) while the current tile feeds the WMMAs.
// ---------------------------------------------------------------------------
template<bool GELU_ACT, bool OUT_BF16, bool ACCUM,
         bool HAS_BIAS, bool HAS_GATE, bool HAS_SCALE>
__global__ __launch_bounds__(256) void gemm_wmma_kernel(
    const __bf16* __restrict__ A, int lda,     // [M,K] row-major
    const __bf16* __restrict__ Bt, int ldbt,   // [N,K] row-major (ldbt == K)
    const float* __restrict__ bias,
    const float* __restrict__ gate, int gateStride,
    const float* __restrict__ scale_ptr,
    float* __restrict__ Cf, __bf16* __restrict__ Cb, int ldc,
    int K)
{
    __shared__ __attribute__((aligned(16))) __bf16 As[2][64][32];    // 2x4 KB
    __shared__ __attribute__((aligned(16))) __bf16 Bs[2][128][32];   // 2x8 KB

    const int tid  = threadIdx.x;
    const int wid  = tid >> 5;
    const int lane = tid & 31;
    const int lr   = lane & 15;   // lane % 16
    const int lh   = lane >> 4;   // lane half
    const int waveM = wid & 3;    // 0..3  -> 16-row groups
    const int waveN = wid >> 2;   // 0..1  -> 64-col groups
    const int rowBase = blockIdx.y * 64;
    const int colBase = blockIdx.x * 128;

    v8f acc[4] = {{}, {}, {}, {}};

    // staging coordinates (contiguous 16B chunks)
    const int ar = tid >> 2, ac = (tid & 3) << 3;   // A: 64 rows x 32 K
    const int bn = tid >> 1, bk = (tid & 1) << 4;   // B: 128 rows x 32 K (2 chunks)
    const int arow  = 16 * waveM + lr;
    const int nbase = 64 * waveN + lr;

    const __bf16* gA = A  + (size_t)(rowBase + ar) * lda  + ac;
    const __bf16* gB = Bt + (size_t)(colBase + bn) * ldbt + bk;

    const int iters = K >> 5;

    // prologue: stage tile 0 into buffer 0
#if USE_ASYNC_LDS
    async_copy16(gA, &As[0][ar][ac]);
    async_copy16(gB, &Bs[0][bn][bk]);
    async_copy16(gB + 8, &Bs[0][bn][bk + 8]);
    __builtin_amdgcn_s_wait_asynccnt(0);
    __syncthreads();
#else
    {
        uint4 ra  = *(const uint4*)gA;
        uint4 rb0 = *(const uint4*)gB;
        uint4 rb1 = *(const uint4*)(gB + 8);
        *(uint4*)&As[0][ar][ac]     = ra;
        *(uint4*)&Bs[0][bn][bk]     = rb0;
        *(uint4*)&Bs[0][bn][bk + 8] = rb1;
    }
    __syncthreads();
#endif

    int p = 0;
    for (int it = 0; it < iters; ++it) {
        const bool has_next = (it + 1) < iters;
        const int  kn = (it + 1) << 5;

#if USE_ASYNC_LDS
        if (has_next) {
            async_copy16(gA + kn,     &As[p ^ 1][ar][ac]);
            async_copy16(gB + kn,     &Bs[p ^ 1][bn][bk]);
            async_copy16(gB + kn + 8, &Bs[p ^ 1][bn][bk + 8]);
        }
#else
        uint4 ra, rb0, rb1;
        if (has_next) {
            ra  = *(const uint4*)(gA + kn);
            rb0 = *(const uint4*)(gB + kn);
            rb1 = *(const uint4*)(gB + kn + 8);
        }
#endif

        // A fragment: element j <-> K = lh*16 + j (lanes 0-15 hold K 0-15, lanes
        // 16-31 hold K 16-31) of row M = 16*waveM + lr. Two b128 LDS loads.
        frag_u a, b0, b1, b2, b3;
        a.q[0]  = *(const uint4*)&As[p][arow][lh * 16];
        a.q[1]  = *(const uint4*)&As[p][arow][lh * 16 + 8];
        // B fragments: element j <-> K = lh*16 + j, N = lr within subtile;
        // contiguous 32B per lane in Bs[n][k] -> two b128 LDS loads each.
        b0.q[0] = *(const uint4*)&Bs[p][nbase][lh * 16];
        b0.q[1] = *(const uint4*)&Bs[p][nbase][lh * 16 + 8];
        b1.q[0] = *(const uint4*)&Bs[p][nbase + 16][lh * 16];
        b1.q[1] = *(const uint4*)&Bs[p][nbase + 16][lh * 16 + 8];
        b2.q[0] = *(const uint4*)&Bs[p][nbase + 32][lh * 16];
        b2.q[1] = *(const uint4*)&Bs[p][nbase + 32][lh * 16 + 8];
        b3.q[0] = *(const uint4*)&Bs[p][nbase + 48][lh * 16];
        b3.q[1] = *(const uint4*)&Bs[p][nbase + 48][lh * 16 + 8];

        acc[0] = __builtin_amdgcn_wmma_f32_16x16x32_bf16(
            false, a.v, false, b0.v, (short)0, acc[0], false, false);
        acc[1] = __builtin_amdgcn_wmma_f32_16x16x32_bf16(
            false, a.v, false, b1.v, (short)0, acc[1], false, false);
        acc[2] = __builtin_amdgcn_wmma_f32_16x16x32_bf16(
            false, a.v, false, b2.v, (short)0, acc[2], false, false);
        acc[3] = __builtin_amdgcn_wmma_f32_16x16x32_bf16(
            false, a.v, false, b3.v, (short)0, acc[3], false, false);

#if USE_ASYNC_LDS
        if (has_next) {
            __builtin_amdgcn_s_wait_asynccnt(0);
            __syncthreads();
        }
#else
        if (has_next) {
            *(uint4*)&As[p ^ 1][ar][ac]     = ra;
            *(uint4*)&Bs[p ^ 1][bn][bk]     = rb0;
            *(uint4*)&Bs[p ^ 1][bn][bk + 8] = rb1;
            __syncthreads();
        }
#endif
        p ^= 1;
    }

    const float scale = HAS_SCALE ? *scale_ptr : 1.0f;
    float garr[8];
    if (HAS_GATE) {
        #pragma unroll
        for (int r = 0; r < 8; ++r)
            garr[r] = gate[(size_t)(rowBase + 16 * waveM + lh * 8 + r) * gateStride];
    }

    #pragma unroll
    for (int tt = 0; tt < 4; ++tt) {
        const int ncol = colBase + 64 * waveN + tt * 16 + lr;
        const float bs = HAS_BIAS ? bias[ncol] : 0.0f;
        #pragma unroll
        for (int r = 0; r < 8; ++r) {
            const int mrow = rowBase + 16 * waveM + lh * 8 + r;  // C layout
            float v = acc[tt][r] + bs;
            if (GELU_ACT) v = gelu_tanh(v);
            if (HAS_GATE) v *= garr[r];
            const size_t idx = (size_t)mrow * ldc + ncol;
            if (OUT_BF16)     Cb[idx]  = f2bf(v);
            else if (ACCUM)   Cf[idx] += scale * v;
            else              Cf[idx]  = scale * v;
        }
    }
}

// ---------------------------------------------------------------------------
// Host launcher
// ---------------------------------------------------------------------------
extern "C" void kernel_launch(void* const* d_in, const int* in_sizes, int n_in,
                              void* d_out, int out_size, void* d_ws, size_t ws_size,
                              hipStream_t stream) {
    const float* hidden   = (const float*)d_in[0];
    const float* raw      = (const float*)d_in[1];
    const float* ln_g     = (const float*)d_in[2];
    const float* ln_b     = (const float*)d_in[3];
    const float* pre_g    = (const float*)d_in[4];
    const float* pre_b    = (const float*)d_in[5];
    const float* feat_w   = (const float*)d_in[6];
    const float* feat_b   = (const float*)d_in[7];
    const float* film_w   = (const float*)d_in[8];
    const float* film_b   = (const float*)d_in[9];
    const float* router_w = (const float*)d_in[10];
    const float* router_b = (const float*)d_in[11];
    const float* w1       = (const float*)d_in[12];
    const float* b1       = (const float*)d_in[13];
    const float* w2       = (const float*)d_in[14];
    const float* b2       = (const float*)d_in[15];
    const float* fc1_w    = (const float*)d_in[16];
    const float* fc1_b    = (const float*)d_in[17];
    const float* fc2_w    = (const float*)d_in[18];
    const float* fc2_b    = (const float*)d_in[19];
    const float* alpha    = (const float*)d_in[20];
    float* out = (float*)d_out;

    // workspace carve-out (256B aligned)
    char* ws = (char*)d_ws;
    size_t off = 0;
    auto carve = [&](size_t bytes) {
        void* p = ws + off;
        off = (off + bytes + 255) & ~(size_t)255;
        return p;
    };
    __bf16* xs_bf   = (__bf16*)carve((size_t)NTOK * DMODEL * 2);
    __bf16* xin_bf  = (__bf16*)carve((size_t)NTOK * DMODEL * 2);
    float*  gate8   = (float*) carve((size_t)NTOK * NEXP * 4);
    __bf16* mid_bf  = (__bf16*)carve((size_t)NTOK * DFF * 2);
    __bf16* H_bf    = (__bf16*)carve((size_t)NTOK * HEXP * 2);
    __bf16* w1t_bf  = (__bf16*)carve((size_t)NEXP * DMODEL * HEXP * 2);  // [e][H,D]
    __bf16* w2t_bf  = (__bf16*)carve((size_t)NEXP * HEXP * DMODEL * 2);  // [e][D,H]
    __bf16* fc1t_bf = (__bf16*)carve((size_t)DMODEL * DFF * 2);          // [DFF,D]
    __bf16* fc2t_bf = (__bf16*)carve((size_t)DFF * DMODEL * 2);          // [D,DFF]

    // 1) prep: LN, FiLM, router gates, out init
    prep_kernel<<<NTOK, 256, 0, stream>>>(
        hidden, raw, ln_g, ln_b, pre_g, pre_b, feat_w, feat_b,
        film_w, film_b, router_w, router_b, b2, alpha,
        xs_bf, xin_bf, gate8, out);

    // 2) weight transpose + f32->bf16 (B operands stored as B^T[N,K])
    {
        dim3 blk(32, 8);
        transpose_cvt_kernel<<<dim3(HEXP / 32, DMODEL / 32, NEXP), blk, 0, stream>>>(
            w1, w1t_bf, DMODEL, HEXP);
        transpose_cvt_kernel<<<dim3(DMODEL / 32, HEXP / 32, NEXP), blk, 0, stream>>>(
            w2, w2t_bf, HEXP, DMODEL);
        transpose_cvt_kernel<<<dim3(DFF / 32, DMODEL / 32, 1), blk, 0, stream>>>(
            fc1_w, fc1t_bf, DMODEL, DFF);
        transpose_cvt_kernel<<<dim3(DMODEL / 32, DFF / 32, 1), blk, 0, stream>>>(
            fc2_w, fc2t_bf, DFF, DMODEL);
    }

    // 3) shared FFN: mid = gelu(xs @ fc1 + fc1_b); out += mid @ fc2 + fc2_b
    {
        dim3 g1(DFF / 128, NTOK / 64);
        gemm_wmma_kernel<true, true, false, true, false, false>
            <<<g1, 256, 0, stream>>>(
            xs_bf, DMODEL, fc1t_bf, DMODEL, fc1_b, nullptr, 0, nullptr,
            nullptr, mid_bf, DFF, DMODEL);
        dim3 g2(DMODEL / 128, NTOK / 64);
        gemm_wmma_kernel<false, false, true, true, false, false>
            <<<g2, 256, 0, stream>>>(
            mid_bf, DFF, fc2t_bf, DFF, fc2_b, nullptr, 0, nullptr,
            out, nullptr, DMODEL, DFF);
    }

    // 4) MoE experts (gate is exactly zero off the top-2, so gate-scaled dense
    //    per-expert compute matches the reference's masked dense combine)
    for (int e = 0; e < NEXP; ++e) {
        dim3 g1(HEXP / 128, NTOK / 64);
        gemm_wmma_kernel<true, true, false, true, true, false>
            <<<g1, 256, 0, stream>>>(
            xin_bf, DMODEL,
            w1t_bf + (size_t)e * DMODEL * HEXP, DMODEL,
            b1 + (size_t)e * HEXP,
            gate8 + e, NEXP,            // per-row gate scale after GELU
            nullptr,
            nullptr, H_bf, HEXP, DMODEL);
        dim3 g2(DMODEL / 128, NTOK / 64);
        gemm_wmma_kernel<false, false, true, false, false, true>
            <<<g2, 256, 0, stream>>>(
            H_bf, HEXP,
            w2t_bf + (size_t)e * HEXP * DMODEL, HEXP,
            nullptr, nullptr, 0,
            alpha,                      // out += alpha * (H @ w2_e)
            out, nullptr, DMODEL, HEXP);
    }
}
